// QGRU2_55911884259442
// MI455X (gfx1250) — matrix operand order repaired and checked by reading, not compiled, not run on hardware
//
#include <hip/hip_runtime.h>
#include <stdint.h>

// MI455X / gfx1250: wave32, WMMA 16x16x32 bf16 path.
#define T_LEN 4096
#define B_SZ  64
#define D_SZ  128
#define M_ROWS (B_SZ * T_LEN)   // 262144, row index m = b*T + t

typedef __attribute__((ext_vector_type(16))) __bf16 v16bf;
typedef __attribute__((ext_vector_type(8)))  float  v8f;

union Frag { unsigned u[8]; v16bf v; };

__device__ __forceinline__ unsigned short bf16_rne(float f) {
  union { float f; unsigned u; } a; a.f = f;
  unsigned u = a.u;
  u += 0x7FFFu + ((u >> 16) & 1u);
  return (unsigned short)(u >> 16);
}
__device__ __forceinline__ unsigned pack_bf16(float lo, float hi) {
  return (unsigned)bf16_rne(lo) | ((unsigned)bf16_rne(hi) << 16);
}
__device__ __forceinline__ float sigmoid_f(float x) { return 1.0f / (1.0f + __expf(-x)); }
__device__ __forceinline__ float tanh_f(float x)    { return 1.0f - 2.0f / (__expf(2.0f * x) + 1.0f); }

// ---------------------------------------------------------------------------
// Kernel 1: encoded[m, d] = sum_w x[b,w,t] * enc_w[d,w] + enc_b[d]  (m = b*T+t)
// Stores encoded as bf16 (WMMA ingest format) and encDec[m] = encoded . dec_w.
// K=32 is too small to feed WMMA efficiently; 2.1 GFLOP on VALU is cheap and
// the x access pattern (t contiguous across lanes) stays fully coalesced.
// ---------------------------------------------------------------------------
__global__ void __launch_bounds__(256) encode_kernel(
    const float* __restrict__ x, const float* __restrict__ enc_w,
    const float* __restrict__ enc_b, const float* __restrict__ dec_w,
    unsigned* __restrict__ encBf, float* __restrict__ encDec)
{
  __shared__ float ew[128 * 33];   // enc_w [d][w], padded
  __shared__ float xs[32 * 17];    // x tile [w][r], padded
  __shared__ float eb[128];
  __shared__ float dw[128];
  __shared__ float red[16 * 17];

  const int i  = threadIdx.x;
  const int m0 = blockIdx.x * 16;      // 16 rows: same b, consecutive t
  const int b  = m0 >> 12;
  const int t0 = m0 & 4095;

#pragma unroll
  for (int j = 0; j < 16; ++j) {       // 4096 enc_w elements, coalesced
    int idx = i + j * 256;
    ew[(idx >> 5) * 33 + (idx & 31)] = enc_w[idx];
  }
  if (i < 128) { eb[i] = enc_b[i]; dw[i] = dec_w[i]; }
#pragma unroll
  for (int j = 0; j < 2; ++j) {        // 512 x elements, coalesced along t
    int e = i + j * 256;
    int w = e >> 4, r = e & 15;
    xs[w * 17 + r] = x[(size_t)(b * 32 + w) * 4096 + t0 + r];
  }
  __syncthreads();

  const int r = i >> 4, c = i & 15, d0 = c * 8;
  float edPart = 0.0f;
  unsigned packed[4];
#pragma unroll
  for (int dd = 0; dd < 8; dd += 2) {
    float acc0 = eb[d0 + dd], acc1 = eb[d0 + dd + 1];
#pragma unroll
    for (int w = 0; w < 32; ++w) {
      float xv = xs[w * 17 + r];                 // broadcast within row group
      acc0 = fmaf(xv, ew[(d0 + dd) * 33 + w], acc0);
      acc1 = fmaf(xv, ew[(d0 + dd + 1) * 33 + w], acc1);
    }
    edPart = fmaf(acc0, dw[d0 + dd], edPart);
    edPart = fmaf(acc1, dw[d0 + dd + 1], edPart);
    packed[dd >> 1] = pack_bf16(acc0, acc1);
  }
  *(uint4*)(encBf + (size_t)(m0 + r) * 64 + c * 4) =
      make_uint4(packed[0], packed[1], packed[2], packed[3]);

  red[r * 17 + c] = edPart;
  __syncthreads();
  if (c == 0) {
    float s = 0.0f;
#pragma unroll
    for (int k = 0; k < 16; ++k) s += red[r * 17 + k];
    encDec[m0 + r] = s;
  }
}

// ---------------------------------------------------------------------------
// Kernel 2: gi = encoded @ w_ih^T + b_ih   (M=262144, N=384, K=128) via WMMA.
// Block = 64 M-rows x 128 N-cols; 8 waves = 4 M-tiles x 2 N-halves of 4 tiles.
// w_ih staged transposed+bf16 in LDS; enc tile staged as-is (already bf16).
// ---------------------------------------------------------------------------
__global__ void __launch_bounds__(256) gi_gemm_kernel(
    const unsigned* __restrict__ encBf, const float* __restrict__ w_ih,
    const float* __restrict__ b_ih, float* __restrict__ gi)
{
  __shared__ unsigned wT[128 * 65];    // [n][kpair], bf16x2, padded (33.3 KB)
  __shared__ unsigned encT[64 * 66];   // [row][kpair], bf16x2, padded (16.9 KB)
  __shared__ float    bih[128];

  const int i  = threadIdx.x;
  const int m0 = blockIdx.x * 64;
  const int nb = blockIdx.y;           // 0..2: 128-wide N block

#pragma unroll
  for (int j = 0; j < 32; ++j) {       // 128 n x 64 kpairs, fp32->bf16 RNE
    int idx = i + j * 256;
    int nl = idx >> 6, kp = idx & 63;
    int n = nb * 128 + nl;
    float lo = w_ih[(size_t)n * 128 + kp * 2];
    float hi = w_ih[(size_t)n * 128 + kp * 2 + 1];
    wT[nl * 65 + kp] = pack_bf16(lo, hi);
  }
#pragma unroll
  for (int j = 0; j < 4; ++j) {        // 1024 uint4s of enc tile, coalesced
    int idx4 = i + j * 256;
    int row = idx4 >> 4, col4 = idx4 & 15;
    uint4 v = *(const uint4*)(encBf + (size_t)(m0 + row) * 64 + col4 * 4);
    encT[row * 66 + col4 * 4 + 0] = v.x;
    encT[row * 66 + col4 * 4 + 1] = v.y;
    encT[row * 66 + col4 * 4 + 2] = v.z;
    encT[row * 66 + col4 * 4 + 3] = v.w;
  }
  if (i < 128) bih[i] = b_ih[nb * 128 + i];
  __syncthreads();

  const int lane  = i & 31;
  const int wv    = i >> 5;
  const int mtile = wv & 3;
  const int nhalf = wv >> 2;
  const int rowA  = mtile * 16 + (lane & 15);   // A layout: lane = M row
  const int half  = lane >> 4;

  v8f acc[4];
#pragma unroll
  for (int j = 0; j < 4; ++j) acc[j] = (v8f)(0.0f);

#pragma unroll
  for (int ks = 0; ks < 4; ++ks) {
    Frag a;
#pragma unroll
    for (int v = 0; v < 8; ++v) {
      int kb = ((v & 3) * 2) + ((v >> 2) * 16) + half * 8;   // ISA A-frag K map
      a.u[v] = encT[rowA * 66 + ((ks * 32 + kb) >> 1)];
    }
#pragma unroll
    for (int j = 0; j < 4; ++j) {
      Frag bf;
      int ntL = nhalf * 4 + j;
#pragma unroll
      for (int v = 0; v < 8; ++v) {               // B layout: lane%16 = N, vgpr = K pair
        int kb = ((v & 3) * 2) + ((v >> 2) * 16) + half * 8;
        bf.u[v] = wT[(ntL * 16 + (lane & 15)) * 65 + ((ks * 32 + kb) >> 1)];
      }
      acc[j] = __builtin_amdgcn_wmma_f32_16x16x32_bf16(
          false, a.v, false, bf.v, (short)0, acc[j], false, false);
    }
  }

#pragma unroll
  for (int j = 0; j < 4; ++j) {
    int nL = (nhalf * 4 + j) * 16 + (lane & 15);  // C layout: lane%16 = N
    int mBase = m0 + mtile * 16 + (lane >> 4) * 8;
#pragma unroll
    for (int v = 0; v < 8; ++v)                    // vgpr v -> M = v (+8 hi half)
      gi[(size_t)(mBase + v) * 384 + nb * 128 + nL] = acc[j][v] + bih[nL];
  }
}

// ---------------------------------------------------------------------------
// Kernel 3: persistent GRU recurrence. 4 blocks x 16 batch rows, 8 waves.
// w_hh bf16 B-fragments live in VGPRs for all 4096 steps; h lives in LDS
// (bf16 for WMMA A, fp32 for the cell update). Writes final [B,T] output.
// ---------------------------------------------------------------------------
__global__ void __launch_bounds__(256) gru_recurrence_kernel(
    const float* __restrict__ gi, const float* __restrict__ w_hh,
    const float* __restrict__ b_hh, const float* __restrict__ dec_w,
    const float* __restrict__ dec_b, const float* __restrict__ scale,
    const float* __restrict__ noise, const float* __restrict__ encDec,
    float* __restrict__ out)
{
  __shared__ unsigned hU[16 * 66];     // h as bf16 pairs (WMMA A source)
  __shared__ float gh[16 * 388];       // h @ w_hh^T (pre-bias)
  __shared__ float hF[16 * 132];       // h fp32
  __shared__ float red[16 * 17];
  __shared__ float bhh[384];
  __shared__ float dwS[128];

  const int i    = threadIdx.x;
  const int lane = i & 31;
  const int wv   = i >> 5;
  const int bt   = blockIdx.x;         // batch tile: rows b = bt*16 .. +15
  const int r = i >> 4, c = i & 15, d0 = c * 8;

  if (i < 128) dwS[i] = dec_w[i];
  for (int j = i; j < 384; j += 256) bhh[j] = b_hh[j];
  for (int j = i; j < 16 * 66;  j += 256) hU[j] = 0u;
  for (int j = i; j < 16 * 132; j += 256) hF[j] = 0.0f;

  const float decb0  = dec_b[0];
  const float scale0 = scale[0];

  // Preload this wave's 3 N-tiles x 4 K-steps of w_hh as bf16 B-fragments
  // (96 VGPRs, resident for the whole time loop -> zero weight traffic/step).
  Frag bfr[3][4];
#pragma unroll
  for (int j = 0; j < 3; ++j) {
    int nt = wv * 3 + j;
#pragma unroll
    for (int ks = 0; ks < 4; ++ks) {
      int n = nt * 16 + (lane & 15);
#pragma unroll
      for (int v = 0; v < 8; ++v) {
        int k = ks * 32 + ((v & 3) * 2) + ((v >> 2) * 16) + (lane >> 4) * 8;
        bfr[j][ks].u[v] = pack_bf16(w_hh[(size_t)n * 128 + k],
                                    w_hh[(size_t)n * 128 + k + 1]);
      }
    }
  }
  __syncthreads();

  const int rowA = lane & 15;
  const int half = lane >> 4;
  const size_t rowBase = (size_t)(bt * 16 + r) * 4096;   // gi/out row for (r)

  for (int t = 0; t < 4096; ++t) {
    // Issue gi loads early; consumed after the barrier (overlaps WMMA).
    const float* gp = gi + (rowBase + t) * 384;
    float4 ir0 = *(const float4*)(gp + d0);
    float4 ir1 = *(const float4*)(gp + d0 + 4);
    float4 iz0 = *(const float4*)(gp + 128 + d0);
    float4 iz1 = *(const float4*)(gp + 128 + d0 + 4);
    float4 in0 = *(const float4*)(gp + 256 + d0);
    float4 in1 = *(const float4*)(gp + 256 + d0 + 4);

    // gh = h @ w_hh^T : 12 WMMA per wave, 3 independent depth-4 chains.
    v8f acc[3];
#pragma unroll
    for (int j = 0; j < 3; ++j) acc[j] = (v8f)(0.0f);
#pragma unroll
    for (int ks = 0; ks < 4; ++ks) {
      Frag a;
#pragma unroll
      for (int v = 0; v < 8; ++v) {
        int kb = ((v & 3) * 2) + ((v >> 2) * 16) + half * 8;
        a.u[v] = hU[rowA * 66 + ((ks * 32 + kb) >> 1)];
      }
#pragma unroll
      for (int j = 0; j < 3; ++j)
        acc[j] = __builtin_amdgcn_wmma_f32_16x16x32_bf16(
            false, a.v, false, bfr[j][ks].v, (short)0, acc[j], false, false);
    }
#pragma unroll
    for (int j = 0; j < 3; ++j) {
      int n = (wv * 3 + j) * 16 + (lane & 15);
      int mrow = (lane >> 4) * 8;
#pragma unroll
      for (int v = 0; v < 8; ++v)
        gh[(mrow + v) * 388 + n] = acc[j][v];
    }
    __syncthreads();                    // gh ready

    // GRU cell for (row r, dims d0..d0+7)
    float irv[8] = {ir0.x, ir0.y, ir0.z, ir0.w, ir1.x, ir1.y, ir1.z, ir1.w};
    float izv[8] = {iz0.x, iz0.y, iz0.z, iz0.w, iz1.x, iz1.y, iz1.z, iz1.w};
    float inv[8] = {in0.x, in0.y, in0.z, in0.w, in1.x, in1.y, in1.z, in1.w};
    float hn[8];
    float resPart = 0.0f;
#pragma unroll
    for (int dd = 0; dd < 8; ++dd) {
      int d = d0 + dd;
      float hr  = gh[r * 388 + d]       + bhh[d];
      float hz  = gh[r * 388 + 128 + d] + bhh[128 + d];
      float hnn = gh[r * 388 + 256 + d] + bhh[256 + d];
      float rg = sigmoid_f(irv[dd] + hr);
      float z  = sigmoid_f(izv[dd] + hz);
      float nn = tanh_f(inv[dd] + rg * hnn);
      float hv = (1.0f - z) * nn + z * hF[r * 132 + d];
      hn[dd] = hv;
      resPart = fmaf(hv, dwS[d], resPart);
    }
#pragma unroll
    for (int dd = 0; dd < 8; ++dd) hF[r * 132 + d0 + dd] = hn[dd];
#pragma unroll
    for (int dd = 0; dd < 8; dd += 2)
      hU[r * 66 + ((d0 + dd) >> 1)] = pack_bf16(hn[dd], hn[dd + 1]);
    red[r * 17 + c] = resPart;
    __syncthreads();                    // h(t+1) + partials ready

    if (c == 0) {                       // decoder + noise + ReLU epilogue
      float s = 0.0f;
#pragma unroll
      for (int k = 0; k < 16; ++k) s += red[r * 17 + k];
      size_t m = rowBase + t;
      float ov = encDec[m] + s + decb0 + scale0 * noise[m];
      out[m] = fmaxf(ov, 0.0f);
    }
  }
}

// ---------------------------------------------------------------------------
extern "C" void kernel_launch(void* const* d_in, const int* in_sizes, int n_in,
                              void* d_out, int out_size, void* d_ws, size_t ws_size,
                              hipStream_t stream) {
  (void)in_sizes; (void)n_in; (void)out_size; (void)ws_size;
  const float* x     = (const float*)d_in[0];
  const float* enc_w = (const float*)d_in[1];
  const float* enc_b = (const float*)d_in[2];
  const float* w_ih  = (const float*)d_in[3];
  const float* w_hh  = (const float*)d_in[4];
  const float* b_ih  = (const float*)d_in[5];
  const float* b_hh  = (const float*)d_in[6];
  const float* dec_w = (const float*)d_in[7];
  const float* dec_b = (const float*)d_in[8];
  const float* scale = (const float*)d_in[9];
  const float* noise = (const float*)d_in[10];
  float* out = (float*)d_out;

  // Workspace layout: enc bf16 (64 MB) | encDec f32 (1 MB) | gi f32 (384 MB)
  char* ws = (char*)d_ws;
  unsigned* encBf = (unsigned*)ws;
  float* encDec   = (float*)(ws + (size_t)67108864);
  float* gi       = (float*)(ws + (size_t)68157440);

  encode_kernel<<<M_ROWS / 16, 256, 0, stream>>>(x, enc_w, enc_b, dec_w, encBf, encDec);
  gi_gemm_kernel<<<dim3(M_ROWS / 64, 3), 256, 0, stream>>>(encBf, w_ih, b_ih, gi);
  gru_recurrence_kernel<<<4, 256, 0, stream>>>(gi, w_hh, b_hh, dec_w, dec_b,
                                               scale, noise, encDec, out);
}
